// AttentionBlock_34428457845281
// MI455X (gfx1250) — matrix operand order repaired and checked by reading, compiled, not run on hardware
//
#include <hip/hip_runtime.h>
#include <math.h>

// ---------------------------------------------------------------------------
// Transformer encoder block for MI455X (gfx1250).
// All GEMMs run through v_wmma_f32_16x16x32_bf16 (bf16 inputs, f32 accum).
// Shapes: B=4, N=2048, HD=1024. All tile dims divide evenly (no edge guards).
// ---------------------------------------------------------------------------

#define HDIM  1024
#define SEQ   2048
#define BATCH 4

typedef __bf16 bf16_t;
typedef __attribute__((ext_vector_type(16))) bf16_t v16bf;
typedef __attribute__((ext_vector_type(8)))  float  v8f;

union BF16Frag { v16bf v; unsigned short u[16]; };

__device__ __forceinline__ unsigned short f2bf(float f) {
  // round-to-nearest-even f32 -> bf16
  unsigned u = __float_as_uint(f);
  u += 0x7FFFu + ((u >> 16) & 1u);
  return (unsigned short)(u >> 16);
}

// ---------------------------------------------------------------------------
// Tiled GEMM:  Out[M,Nc] = act( scale * (A[M,K] @ W) + bias )
//   bTrans==0 : W is [K,Nc] row-major
//   bTrans==1 : W is [Nc,K] row-major (i.e. Out = A @ W^T)  -> used for Q@K^T
//   act==1    : exact GELU
// Block: 256 thr = 8 waves; tile 128x128, K-step 32 (one WMMA K per step).
// Wave grid 4x2; each wave computes 32x64 = 2x4 accum tiles of 16x16.
// ---------------------------------------------------------------------------
#define TM 128
#define TN 128
#define TK 32
#define LDP (TK + 2)   // ushort row stride pad

__global__ __launch_bounds__(256) void gemm_bf16_wmma(
    const float* __restrict__ A, const float* __restrict__ Wm,
    const float* __restrict__ bias, float* __restrict__ Out,
    int M, int Nc, int K,
    long long strideA, long long strideW, long long strideO,
    int bTrans, int act, float scale)
{
  (void)M;
  __shared__ unsigned short As[TM][LDP];   // [row][k]    bf16 bits
  __shared__ unsigned short Ws[TN][LDP];   // [col][k]    (W stored transposed)

  const int tid  = threadIdx.x;
  const int lane = tid & 31;
  const int wave = tid >> 5;
  const int wM   = wave >> 1;      // 0..3
  const int wN   = wave & 1;       // 0..1
  const int half = lane >> 4;      // 0/1 (lane group)
  const int l16  = lane & 15;

  const int bm = blockIdx.y * TM;
  const int bn = blockIdx.x * TN;

  const float* Ab = A   + (long long)blockIdx.z * strideA;
  const float* Wb = Wm  + (long long)blockIdx.z * strideW;
  float*       Ob = Out + (long long)blockIdx.z * strideO;

  v8f acc[2][4];
  #pragma unroll
  for (int i = 0; i < 2; ++i)
    #pragma unroll
    for (int j = 0; j < 4; ++j)
      acc[i][j] = (v8f){0.f,0.f,0.f,0.f,0.f,0.f,0.f,0.f};

  for (int k0 = 0; k0 < K; k0 += TK) {
    // ---- stage A tile (128x32 f32 -> bf16), 4 float4 per thread ----
    #pragma unroll
    for (int it = 0; it < 4; ++it) {
      const int f   = tid + it * 256;
      const int row = f >> 3;
      const int c4  = (f & 7) << 2;
      const float4 d = *(const float4*)(Ab + (size_t)(bm + row) * K + k0 + c4);
      As[row][c4 + 0] = f2bf(d.x);
      As[row][c4 + 1] = f2bf(d.y);
      As[row][c4 + 2] = f2bf(d.z);
      As[row][c4 + 3] = f2bf(d.w);
      if (k0 + TK < K)  // prefetch next K tile (global_prefetch_b8)
        __builtin_prefetch(Ab + (size_t)(bm + row) * K + k0 + TK + c4, 0, 3);
    }
    // ---- stage W tile transposed into LDS: Ws[col][k] ----
    if (!bTrans) {
      #pragma unroll
      for (int it = 0; it < 4; ++it) {
        const int f  = tid + it * 256;
        const int kr = f >> 5;            // 0..31
        const int c4 = (f & 31) << 2;     // 0..124
        const float4 d = *(const float4*)(Wb + (size_t)(k0 + kr) * Nc + bn + c4);
        Ws[c4 + 0][kr] = f2bf(d.x);
        Ws[c4 + 1][kr] = f2bf(d.y);
        Ws[c4 + 2][kr] = f2bf(d.z);
        Ws[c4 + 3][kr] = f2bf(d.w);
      }
    } else {
      #pragma unroll
      for (int it = 0; it < 4; ++it) {
        const int f  = tid + it * 256;
        const int nr = f >> 3;            // 0..127
        const int c4 = (f & 7) << 2;      // 0..28
        const float4 d = *(const float4*)(Wb + (size_t)(bn + nr) * K + k0 + c4);
        Ws[nr][c4 + 0] = f2bf(d.x);
        Ws[nr][c4 + 1] = f2bf(d.y);
        Ws[nr][c4 + 2] = f2bf(d.z);
        Ws[nr][c4 + 3] = f2bf(d.w);
      }
    }
    __syncthreads();

    // ---- A fragments: ISA 16-bit A 16x32 lane/VGPR layout ----
    BF16Frag afr[2];
    #pragma unroll
    for (int ti = 0; ti < 2; ++ti) {
      const int m = wM * 32 + ti * 16 + l16;
      #pragma unroll
      for (int e = 0; e < 16; ++e) {
        const int j  = e >> 1;
        const int r  = e & 1;
        const int kk = ((j >> 2) << 4) + (half << 3) + ((j & 3) << 1) + r;
        afr[ti].u[e] = As[m][kk];
      }
    }
    // ---- B fragments: lane n holds K half*8..+7 and 16+half*8..+7 ----
    BF16Frag bfr[4];
    #pragma unroll
    for (int tj = 0; tj < 4; ++tj) {
      const int n = wN * 64 + tj * 16 + l16;
      #pragma unroll
      for (int e = 0; e < 16; ++e)
        bfr[tj].u[e] = Ws[n][((e >> 3) << 4) + (half << 3) + (e & 7)];
    }
    // ---- 8 WMMAs per wave per K tile ----
    #pragma unroll
    for (int ti = 0; ti < 2; ++ti)
      #pragma unroll
      for (int tj = 0; tj < 4; ++tj)
        acc[ti][tj] = __builtin_amdgcn_wmma_f32_16x16x32_bf16(
            false, afr[ti].v, false, bfr[tj].v, (short)0, acc[ti][tj],
            false, false);

    __syncthreads();
  }

  // ---- epilogue: C/D layout (VGPR v -> row v + 8*half, lane%16 -> col) ----
  #pragma unroll
  for (int ti = 0; ti < 2; ++ti) {
    #pragma unroll
    for (int tj = 0; tj < 4; ++tj) {
      const int   col = bn + wN * 64 + tj * 16 + l16;
      const float bv  = bias ? bias[col] : 0.0f;
      #pragma unroll
      for (int vv = 0; vv < 8; ++vv) {
        const int row = bm + wM * 32 + ti * 16 + (half << 3) + vv;
        float r = acc[ti][tj][vv] * scale + bv;
        if (act == 1) r = 0.5f * r * (1.0f + erff(r * 0.70710678118654752f));
        Ob[(size_t)row * Nc + col] = r;
      }
    }
  }
}

// ---------------------------------------------------------------------------
// Row softmax over ncols (one block per row, 256 threads, wave32 reductions)
// ---------------------------------------------------------------------------
__global__ __launch_bounds__(256) void softmax_rows(float* __restrict__ S, int ncols)
{
  const size_t base = (size_t)blockIdx.x * (size_t)ncols;
  const int tid = threadIdx.x;
  __shared__ float red[8];
  __shared__ float bcast;

  float m = -3.402823e38f;
  for (int c = tid; c < ncols; c += 256) m = fmaxf(m, S[base + c]);
  #pragma unroll
  for (int o = 16; o > 0; o >>= 1) m = fmaxf(m, __shfl_xor(m, o, 32));
  if ((tid & 31) == 0) red[tid >> 5] = m;
  __syncthreads();
  if (tid == 0) {
    float mm = red[0];
    for (int i = 1; i < 8; ++i) mm = fmaxf(mm, red[i]);
    bcast = mm;
  }
  __syncthreads();
  m = bcast;

  float s = 0.0f;
  for (int c = tid; c < ncols; c += 256) {
    const float e = __expf(S[base + c] - m);
    S[base + c] = e;
    s += e;
  }
  #pragma unroll
  for (int o = 16; o > 0; o >>= 1) s += __shfl_xor(s, o, 32);
  __syncthreads();
  if ((tid & 31) == 0) red[tid >> 5] = s;
  __syncthreads();
  if (tid == 0) {
    float ss = 0.0f;
    for (int i = 0; i < 8; ++i) ss += red[i];
    bcast = ss;
  }
  __syncthreads();
  const float inv = 1.0f / bcast;
  for (int c = tid; c < ncols; c += 256) S[base + c] *= inv;
}

// ---------------------------------------------------------------------------
// out = LayerNorm(res + y) * g + b    (row = HD=1024, block = 256 thr x 4 el)
// ---------------------------------------------------------------------------
__global__ __launch_bounds__(256) void add_layernorm(
    const float* __restrict__ res, const float* __restrict__ y,
    const float* __restrict__ g,   const float* __restrict__ b,
    float* __restrict__ out)
{
  const size_t base = (size_t)blockIdx.x * (size_t)HDIM;
  const int tid = threadIdx.x;
  const int c0  = tid << 2;
  __shared__ float red[8];
  __shared__ float bc0, bc1;

  const float4 r4 = *(const float4*)(res + base + c0);
  const float4 y4 = *(const float4*)(y   + base + c0);
  float v0 = r4.x + y4.x, v1 = r4.y + y4.y, v2 = r4.z + y4.z, v3 = r4.w + y4.w;

  float s = v0 + v1 + v2 + v3;
  #pragma unroll
  for (int o = 16; o > 0; o >>= 1) s += __shfl_xor(s, o, 32);
  if ((tid & 31) == 0) red[tid >> 5] = s;
  __syncthreads();
  if (tid == 0) {
    float t = 0.0f;
    for (int i = 0; i < 8; ++i) t += red[i];
    bc0 = t * (1.0f / (float)HDIM);
  }
  __syncthreads();
  const float mu = bc0;

  const float d0 = v0 - mu, d1 = v1 - mu, d2 = v2 - mu, d3 = v3 - mu;
  float vs = d0 * d0 + d1 * d1 + d2 * d2 + d3 * d3;
  #pragma unroll
  for (int o = 16; o > 0; o >>= 1) vs += __shfl_xor(vs, o, 32);
  __syncthreads();
  if ((tid & 31) == 0) red[tid >> 5] = vs;
  __syncthreads();
  if (tid == 0) {
    float t = 0.0f;
    for (int i = 0; i < 8; ++i) t += red[i];
    bc1 = t * (1.0f / (float)HDIM);
  }
  __syncthreads();
  const float inv = rsqrtf(bc1 + 1e-5f);

  out[base + c0 + 0] = d0 * inv * g[c0 + 0] + b[c0 + 0];
  out[base + c0 + 1] = d1 * inv * g[c0 + 1] + b[c0 + 1];
  out[base + c0 + 2] = d2 * inv * g[c0 + 2] + b[c0 + 2];
  out[base + c0 + 3] = d3 * inv * g[c0 + 3] + b[c0 + 3];
}

// ---------------------------------------------------------------------------
extern "C" void kernel_launch(void* const* d_in, const int* in_sizes, int n_in,
                              void* d_out, int out_size, void* d_ws, size_t ws_size,
                              hipStream_t stream)
{
  (void)in_sizes; (void)n_in; (void)out_size; (void)ws_size;

  const float* x   = (const float*)d_in[0];
  const float* Wq  = (const float*)d_in[1];  const float* bq  = (const float*)d_in[2];
  const float* Wk  = (const float*)d_in[3];  const float* bk  = (const float*)d_in[4];
  const float* Wv  = (const float*)d_in[5];  const float* bv  = (const float*)d_in[6];
  const float* Wo  = (const float*)d_in[7];  const float* bo  = (const float*)d_in[8];
  const float* g1  = (const float*)d_in[9];  const float* b1  = (const float*)d_in[10];
  const float* g2  = (const float*)d_in[11]; const float* b2  = (const float*)d_in[12];
  const float* Wf1 = (const float*)d_in[13]; const float* bf1 = (const float*)d_in[14];
  const float* Wf2 = (const float*)d_in[15]; const float* bf2 = (const float*)d_in[16];

  const int M  = BATCH * SEQ;                  // 8192 token rows
  const size_t xsz = (size_t)M * HDIM;         // 8M f32
  const size_t ssz = (size_t)BATCH * SEQ * SEQ;// 16M f32

  float* ws  = (float*)d_ws;
  float* q   = ws;                 // [M, HD]
  float* kx  = q  + xsz;           // [M, HD]
  float* vx  = kx + xsz;           // [M, HD]
  float* S   = vx + xsz;           // [B, N, N] scores/attn (reused as o-proj out)
  float* t   = S  + ssz;           // [M, HD] attn output
  float* x1  = t  + xsz;           // [M, HD] after LN1
  float* h   = x1 + xsz;           // [M, 4*HD] FFN hidden
  float* f2  = h  + (size_t)M * 4 * HDIM;      // [M, HD]

  const dim3 blk(256);

  // --- QKV projections: [8192,1024] @ [1024,1024] + b ---
  {
    dim3 grd(HDIM / TN, M / TM, 1);
    gemm_bf16_wmma<<<grd, blk, 0, stream>>>(x, Wq, bq, q,  M, HDIM, HDIM, 0, 0, 0, 0, 0, 1.0f);
    gemm_bf16_wmma<<<grd, blk, 0, stream>>>(x, Wk, bk, kx, M, HDIM, HDIM, 0, 0, 0, 0, 0, 1.0f);
    gemm_bf16_wmma<<<grd, blk, 0, stream>>>(x, Wv, bv, vx, M, HDIM, HDIM, 0, 0, 0, 0, 0, 1.0f);
  }

  // --- scores: S_b = (Q_b @ K_b^T) / sqrt(HD), batched over B ---
  {
    dim3 grd(SEQ / TN, SEQ / TM, BATCH);
    gemm_bf16_wmma<<<grd, blk, 0, stream>>>(q, kx, nullptr, S,
        SEQ, SEQ, HDIM,
        (long long)SEQ * HDIM, (long long)SEQ * HDIM, (long long)SEQ * SEQ,
        /*bTrans=*/1, 0, 0.03125f /* 1/sqrt(1024) */);
  }

  // --- softmax over each of B*N rows of length N ---
  softmax_rows<<<dim3(BATCH * SEQ), blk, 0, stream>>>(S, SEQ);

  // --- t_b = P_b @ V_b, batched ---
  {
    dim3 grd(HDIM / TN, SEQ / TM, BATCH);
    gemm_bf16_wmma<<<grd, blk, 0, stream>>>(S, vx, nullptr, t,
        SEQ, HDIM, SEQ,
        (long long)SEQ * SEQ, (long long)SEQ * HDIM, (long long)SEQ * HDIM,
        0, 0, 1.0f);
  }

  // --- output projection (reuse S as [M, HD] buffer) ---
  float* o = S;
  {
    dim3 grd(HDIM / TN, M / TM, 1);
    gemm_bf16_wmma<<<grd, blk, 0, stream>>>(t, Wo, bo, o, M, HDIM, HDIM, 0, 0, 0, 0, 0, 1.0f);
  }

  // --- x1 = LN(x + o) ---
  add_layernorm<<<dim3(M), blk, 0, stream>>>(x, o, g1, b1, x1);

  // --- FFN1 with exact GELU: h = gelu(x1 @ Wf1 + bf1), [8192,4096] ---
  {
    dim3 grd((4 * HDIM) / TN, M / TM, 1);
    gemm_bf16_wmma<<<grd, blk, 0, stream>>>(x1, Wf1, bf1, h, M, 4 * HDIM, HDIM, 0, 0, 0, 0, /*act=*/1, 1.0f);
  }

  // --- FFN2: f2 = h @ Wf2 + bf2 ---
  {
    dim3 grd(HDIM / TN, M / TM, 1);
    gemm_bf16_wmma<<<grd, blk, 0, stream>>>(h, Wf2, bf2, f2, M, HDIM, 4 * HDIM, 0, 0, 0, 0, 0, 1.0f);
  }

  // --- final: out = LN(x1 + f2) ---
  add_layernorm<<<dim3(M), blk, 0, stream>>>(x1, f2, g2, b2, (float*)d_out);
}
